// GNNEncoder_14388140441815
// MI455X (gfx1250) — compile-verified, matrix operand order checked by reading, not run on hardware
//
#include <hip/hip_runtime.h>

// -------- CDNA5 WMMA types --------
typedef __attribute__((ext_vector_type(16))) __bf16 v16bf;
typedef __attribute__((ext_vector_type(8)))  float  v8f;

#define IN_C  128
#define HID_C 128
#define OUT_C 64

static __device__ __forceinline__ __bf16 f2bf(float f) { return (__bf16)f; }

// K-index mapping for the 16-bit A/B operand of V_WMMA_*_16X16X32 (ISA 7.12.2):
// lanes 0-15 : slots 0..7 -> K=0..7,  slots 8..15 -> K=16..23
// lanes 16-31: slots 0..7 -> K=8..15, slots 8..15 -> K=24..31
static __device__ __forceinline__ int kslot(int i, int hi) {
    return ((i < 8) ? i : (i + 8)) + hi * 8;
}

// ---------------- degree / normalization ----------------
__global__ void deg_zero(float* deg, int N) {
    int i = blockIdx.x * blockDim.x + threadIdx.x;
    if (i < N) deg[i] = 0.0f;
}

__global__ void deg_count(const int* __restrict__ dst, float* __restrict__ deg, int E) {
    int e = blockIdx.x * blockDim.x + threadIdx.x;
    if (e < E) atomicAdd(&deg[dst[e]], 1.0f);
}

__global__ void deg_finalize(float* dinv, int N) {
    int i = blockIdx.x * blockDim.x + threadIdx.x;
    if (i < N) dinv[i] = rsqrtf(dinv[i] + 1.0f);   // +1 self-loop, deg >= 1 -> safe
}

// ---------------- pack weights (fp32 row-major K x NC) into bf16 WMMA-B layout ----
__global__ void pack_w(const float* __restrict__ W, __bf16* __restrict__ out, int K, int NC) {
    int t      = blockIdx.x * blockDim.x + threadIdx.x;
    int lane   = t & 31;
    int grp    = t >> 5;                       // tile_n * ksteps + ks
    int ksteps = K >> 5;
    if (grp >= (NC >> 4) * ksteps) return;
    int tile_n = grp / ksteps;
    int ks     = grp - tile_n * ksteps;
    int hi = lane >> 4, ln = lane & 15;
    __bf16* o = out + ((size_t)grp * 32 + lane) * 16;
#pragma unroll
    for (int i = 0; i < 16; ++i) {
        int k = ks * 32 + kslot(i, hi);
        o[i] = f2bf(W[(size_t)k * NC + tile_n * 16 + ln]);
    }
}

// ---------------- GEMM1: h0 = x @ W1 ; agg1 = h0 * dinv^2 ----------------
// One wave owns a 16-row band; A held in registers, reused across all 8 tile_n.
__global__ void gemm1_wmma(const float* __restrict__ x,
                           const __bf16* __restrict__ W1p,
                           const float* __restrict__ dinv,
                           float* __restrict__ h0,
                           float* __restrict__ agg1,
                           int N) {
    int wave = (blockIdx.x * blockDim.x + threadIdx.x) >> 5;
    int lane = threadIdx.x & 31;
    if (wave * 16 >= N) return;                // wave-uniform, EXEC stays all-ones
    bool full = (wave + 1) * 16 <= N;          // wave-uniform: whole band in-bounds

    int hi = lane >> 4;
    int ln = lane & 15;
    int arow = wave * 16 + ln;
    if (arow >= N) arow = 0;                   // clamped load (stores are guarded)
    const float* Ar = x + (size_t)arow * IN_C;

    // Load entire A band (K=128) as bf16 operands: 4 K-steps x v16bf
    v16bf a[4];
#pragma unroll
    for (int ks = 0; ks < 4; ++ks)
#pragma unroll
        for (int i = 0; i < 16; ++i)
            a[ks][i] = f2bf(Ar[ks * 32 + kslot(i, hi)]);

    // Per-lane output rows and their dinv^2 (hoisted out of the tile_n loop)
    int   row0 = wave * 16 + hi * 8;
    float d2[8];
#pragma unroll
    for (int r = 0; r < 8; ++r) {
        int orow = row0 + r;
        float di = dinv[orow < N ? orow : 0];
        d2[r] = di * di;
    }

    const v16bf* Bp = (const v16bf*)W1p;
#pragma unroll
    for (int tn = 0; tn < 8; ++tn) {
        v8f c = {0.f, 0.f, 0.f, 0.f, 0.f, 0.f, 0.f, 0.f};
#pragma unroll
        for (int ks = 0; ks < 4; ++ks) {
            v16bf b = Bp[((size_t)tn * 4 + ks) * 32 + lane];   // 32B contiguous/lane
            c = __builtin_amdgcn_wmma_f32_16x16x32_bf16(false, a[ks], false, b,
                                                        (short)0, c, false, false);
        }
        int col = tn * 16 + ln;
        float* hp = h0   + (size_t)row0 * HID_C + col;
        float* ap = agg1 + (size_t)row0 * HID_C + col;
        if (full) {                            // common path: branch-free stores
#pragma unroll
            for (int r = 0; r < 8; ++r) {
                hp[(size_t)r * HID_C] = c[r];
                ap[(size_t)r * HID_C] = c[r] * d2[r];
            }
        } else {                               // boundary wave only
#pragma unroll
            for (int r = 0; r < 8; ++r) {
                if (row0 + r < N) {
                    hp[(size_t)r * HID_C] = c[r];
                    ap[(size_t)r * HID_C] = c[r] * d2[r];
                }
            }
        }
    }
}

// ---------------- edge scatter, layer 1 (128 ch, 4 per lane) ----------------
__global__ void scatter1(const int* __restrict__ src, const int* __restrict__ dst,
                         const float* __restrict__ dinv,
                         const float* __restrict__ h0,
                         float* __restrict__ agg1, int E) {
    int e    = (blockIdx.x * blockDim.x + threadIdx.x) >> 5;
    int lane = threadIdx.x & 31;
    if (e >= E) return;
    int s = src[e], d = dst[e];
    float nrm = dinv[s] * dinv[d];
    float4 v = ((const float4*)(h0 + (size_t)s * HID_C))[lane];
    float* ap = agg1 + (size_t)d * HID_C + lane * 4;
    atomicAdd(ap + 0, v.x * nrm);
    atomicAdd(ap + 1, v.y * nrm);
    atomicAdd(ap + 2, v.z * nrm);
    atomicAdd(ap + 3, v.w * nrm);
}

// ---------------- GEMM2: h2 = relu(agg1 + b1) @ W2 ; out = h2*dinv^2 + b2 ----
__global__ void gemm2_wmma(const float* __restrict__ agg1,
                           const float* __restrict__ b1,
                           const __bf16* __restrict__ W2p,
                           const float* __restrict__ b2,
                           const float* __restrict__ dinv,
                           float* __restrict__ h2,
                           float* __restrict__ out,
                           int N) {
    int wave = (blockIdx.x * blockDim.x + threadIdx.x) >> 5;
    int lane = threadIdx.x & 31;
    if (wave * 16 >= N) return;                // wave-uniform
    bool full = (wave + 1) * 16 <= N;

    int hi = lane >> 4;
    int ln = lane & 15;
    int arow = wave * 16 + ln;
    if (arow >= N) arow = 0;
    const float* Ar = agg1 + (size_t)arow * HID_C;

    // A = relu(agg1 + b1), converted once, reused for all 4 tile_n
    v16bf a[4];
#pragma unroll
    for (int ks = 0; ks < 4; ++ks)
#pragma unroll
        for (int i = 0; i < 16; ++i) {
            int k = ks * 32 + kslot(i, hi);
            float av = Ar[k] + b1[k];
            a[ks][i] = f2bf(av > 0.0f ? av : 0.0f);
        }

    int   row0 = wave * 16 + hi * 8;
    float d2[8];
#pragma unroll
    for (int r = 0; r < 8; ++r) {
        int orow = row0 + r;
        float di = dinv[orow < N ? orow : 0];
        d2[r] = di * di;
    }

    const v16bf* Bp = (const v16bf*)W2p;
#pragma unroll
    for (int tn = 0; tn < 4; ++tn) {
        v8f c = {0.f, 0.f, 0.f, 0.f, 0.f, 0.f, 0.f, 0.f};
#pragma unroll
        for (int ks = 0; ks < 4; ++ks) {
            v16bf b = Bp[((size_t)tn * 4 + ks) * 32 + lane];
            c = __builtin_amdgcn_wmma_f32_16x16x32_bf16(false, a[ks], false, b,
                                                        (short)0, c, false, false);
        }
        int col    = tn * 16 + ln;
        float bias = b2[col];
        float* hp  = h2  + (size_t)row0 * OUT_C + col;
        float* op  = out + (size_t)row0 * OUT_C + col;
        if (full) {
#pragma unroll
            for (int r = 0; r < 8; ++r) {
                hp[(size_t)r * OUT_C] = c[r];
                op[(size_t)r * OUT_C] = c[r] * d2[r] + bias;
            }
        } else {
#pragma unroll
            for (int r = 0; r < 8; ++r) {
                if (row0 + r < N) {
                    hp[(size_t)r * OUT_C] = c[r];
                    op[(size_t)r * OUT_C] = c[r] * d2[r] + bias;
                }
            }
        }
    }
}

// ---------------- edge scatter, layer 2 (64 ch, 2 per lane) ----------------
__global__ void scatter2(const int* __restrict__ src, const int* __restrict__ dst,
                         const float* __restrict__ dinv,
                         const float* __restrict__ h2,
                         float* __restrict__ out, int E) {
    int e    = (blockIdx.x * blockDim.x + threadIdx.x) >> 5;
    int lane = threadIdx.x & 31;
    if (e >= E) return;
    int s = src[e], d = dst[e];
    float nrm = dinv[s] * dinv[d];
    float2 v = ((const float2*)(h2 + (size_t)s * OUT_C))[lane];
    float* op = out + (size_t)d * OUT_C + lane * 2;
    atomicAdd(op + 0, v.x * nrm);
    atomicAdd(op + 1, v.y * nrm);
}

extern "C" void kernel_launch(void* const* d_in, const int* in_sizes, int n_in,
                              void* d_out, int out_size, void* d_ws, size_t ws_size,
                              hipStream_t stream) {
    const float* x    = (const float*)d_in[0];
    const int*   edge = (const int*)  d_in[1];
    const float* W1   = (const float*)d_in[2];
    const float* b1   = (const float*)d_in[3];
    const float* W2   = (const float*)d_in[4];
    const float* b2   = (const float*)d_in[5];
    float*       out  = (float*)d_out;

    const int N = in_sizes[0] / IN_C;       // 100000
    const int E = in_sizes[1] / 2;          // 1600000
    const int* src = edge;
    const int* dst = edge + E;

    // ws layout: dinv[Na] | W1p bf16[128*128] | W2p bf16[128*64] | h0[N*128] | agg1[N*128]
    float*  ws   = (float*)d_ws;
    size_t  Na   = ((size_t)N + 255) & ~(size_t)255;
    float*  dinv = ws;
    __bf16* W1p  = (__bf16*)(ws + Na);
    __bf16* W2p  = W1p + (size_t)IN_C * HID_C;
    float*  h0   = (float*)(W2p + (size_t)HID_C * OUT_C);  // offset stays 1KB-aligned
    float*  agg1 = h0 + (size_t)N * HID_C;
    float*  h2   = h0;                       // safe: h0 last read by scatter1

    const int TB = 256;
    deg_zero    <<<(N + TB - 1) / TB, TB, 0, stream>>>(dinv, N);
    deg_count   <<<(E + TB - 1) / TB, TB, 0, stream>>>(dst, dinv, E);
    deg_finalize<<<(N + TB - 1) / TB, TB, 0, stream>>>(dinv, N);

    // pack weights into WMMA-B bf16 layout (tiny, one-shot per call)
    pack_w<<<(8 * 4 * 32 + TB - 1) / TB, TB, 0, stream>>>(W1, W1p, IN_C, HID_C);
    pack_w<<<(4 * 4 * 32 + TB - 1) / TB, TB, 0, stream>>>(W2, W2p, HID_C, OUT_C);

    const int tilesM = (N + 15) / 16;        // 6250 row bands, one wave each
    gemm1_wmma<<<(tilesM * 32 + TB - 1) / TB, TB, 0, stream>>>(x, W1p, dinv, h0, agg1, N);
    scatter1  <<<(E + 7) / 8, TB, 0, stream>>>(src, dst, dinv, h0, agg1, E);
    gemm2_wmma<<<(tilesM * 32 + TB - 1) / TB, TB, 0, stream>>>(agg1, b1, W2p, b2, dinv, h2, out, N);
    scatter2  <<<(E + 7) / 8, TB, 0, stream>>>(src, dst, dinv, h2, out, E);
}
